// FeatureFusionModel_64407329571195
// MI455X (gfx1250) — matrix operand order repaired and checked by reading, not compiled, not run on hardware
//
#include <hip/hip_runtime.h>
#include <hip/hip_bf16.h>
#include <stdint.h>

// ---------------- problem constants (from reference) ----------------
#define BB     2
#define VV     65536
#define NVIEW  6
#define DD     384
#define MM     1369      // 37*37 patch tokens
#define PF     64
#define FIN    448       // PF + DD
#define HID    256
#define NOUT   16
#define GRIDP  37
#define KS1    14        // 448 / 32
#define NT1    16        // 256 / 16
#define KS2    8         // 256 / 32

typedef __attribute__((ext_vector_type(16))) __bf16          v16bf;
typedef __attribute__((ext_vector_type(8)))  float           v8f;
typedef __attribute__((ext_vector_type(16))) unsigned short  v16u;
typedef __attribute__((ext_vector_type(8)))  unsigned short  v8u;

union BCast16 { v16u u; v16bf b; };

static __device__ __forceinline__ unsigned short f2bf(float f) {
    unsigned u = __float_as_uint(f);
    u += 0x7FFFu + ((u >> 16) & 1u);      // round-to-nearest-even
    return (unsigned short)(u >> 16);
}

// ---------------------------------------------------------------
// Pre-pass: swizzle W1 (448x256 f32) into WMMA B-fragment order,
// bf16.  Layout: frag[((ks*NT1 + t)*32 + lane)*16 + e]
// B-matrix 32x16 (16-bit) per ISA: lane holds column n = lane&15,
// element e -> k = 8*(lane>>4) + e + (e>=8 ? 8 : 0)  (within this 32-deep K step).
// ---------------------------------------------------------------
__global__ void build_w1frag(const float* __restrict__ W1,
                             unsigned short* __restrict__ w1f) {
    int tid = blockIdx.x * blockDim.x + threadIdx.x;
    if (tid >= KS1 * NT1 * 32 * 16) return;
    int e    = tid & 15;
    int lane = (tid >> 4) & 31;
    int t    = (tid >> 9) & 15;
    int ks   = tid >> 13;
    int k = 32 * ks + 8 * (lane >> 4) + e + (e >= 8 ? 8 : 0);
    int n = 16 * t + (lane & 15);
    w1f[tid] = f2bf(W1[k * HID + n]);
}

__global__ void build_w2frag(const float* __restrict__ W2,
                             unsigned short* __restrict__ w2f) {
    int tid = blockIdx.x * blockDim.x + threadIdx.x;
    if (tid >= KS2 * 32 * 16) return;
    int e    = tid & 15;
    int lane = (tid >> 4) & 31;
    int ks   = tid >> 9;
    int k = 32 * ks + 8 * (lane >> 4) + e + (e >= 8 ? 8 : 0);
    int n = lane & 15;
    w2f[tid] = f2bf(W2[k * NOUT + n]);
}

// ---------------------------------------------------------------
// Fused: projection -> gather/mean -> MLP (bf16 WMMA, f32 accum)
// 128 threads = 4 waves, each wave owns 16 rows (voxels).
// ---------------------------------------------------------------
__global__ __launch_bounds__(128)
void fused_fusion_mlp(const float* __restrict__ patch_tokens,
                      const float* __restrict__ voxel_features,
                      const float* __restrict__ voxel_coords,
                      const float* __restrict__ Kmat,
                      const float* __restrict__ Rt,
                      const unsigned short* __restrict__ w1f,
                      const float* __restrict__ b1,
                      const unsigned short* __restrict__ w2f,
                      const float* __restrict__ b2,
                      float* __restrict__ out) {
    __shared__ __align__(16) unsigned short smem[4][16 * FIN];

    const int lane  = threadIdx.x & 31;
    const int wid   = threadIdx.x >> 5;
    const int khalf = lane >> 4;     // which half-wave (ISA K split)
    const int nlo   = lane & 15;     // row (A) / column (B,C,D) id
    unsigned short* rowbuf = smem[wid];

    const int rowbase = blockIdx.x * 64 + wid * 16;   // first global row

    // ---- projection for row (rowbase + nlo); both halves redundant ----
    int myrow = rowbase + nlo;
    int b     = myrow >> 16;          // V = 65536
    float x = voxel_coords[(size_t)myrow * 3 + 0];
    float y = voxel_coords[(size_t)myrow * 3 + 1];
    float z = voxel_coords[(size_t)myrow * 3 + 2];
    const float* rt = Rt + b * 16;
    float c0 = rt[0] * x + rt[1] * y + rt[2]  * z + rt[3];
    float c1 = rt[4] * x + rt[5] * y + rt[6]  * z + rt[7];
    float c2 = rt[8] * x + rt[9] * y + rt[10] * z + rt[11];
    const float* km = Kmat + b * 9;
    float p0 = km[0] * c0 + km[1] * c1 + km[2] * c2;
    float p1 = km[3] * c0 + km[4] * c1 + km[5] * c2;
    float p2 = km[6] * c0 + km[7] * c1 + km[8] * c2;
    float inv = 1.0f / (p2 + 1e-12f);
    float px = p0 * inv * (518.0f / 1600.0f) * (1.0f / 14.0f);
    float py = p1 * inv * (518.0f /  900.0f) * (1.0f / 14.0f);
    px = fminf(fmaxf(px, 0.0f), (float)(GRIDP - 1));
    py = fminf(fmaxf(py, 0.0f), (float)(GRIDP - 1));
    int idx = (int)px * GRIDP + (int)py;

    // ---- stage 16 fused rows (bf16) into LDS: [64 vox feat | 384 img mean] ----
    for (int r = 0; r < 16; ++r) {
        int fidx = __shfl(idx, r, 32);
        int rrow = rowbase + r;
        int rb   = rrow >> 16;
        const float* pt = patch_tokens +
            ((size_t)rb * NVIEW * MM + (size_t)fidx) * DD;
        const float* vf = voxel_features + (size_t)rrow * PF;
        for (int c = lane; c < FIN; c += 32) {
            float val;
            if (c < PF) {
                val = vf[c];
            } else {
                int d = c - PF;
                float s = 0.0f;
#pragma unroll
                for (int view = 0; view < NVIEW; ++view)
                    s += pt[(size_t)view * MM * DD + d];
                val = s * (1.0f / 6.0f);
            }
            rowbuf[r * FIN + c] = f2bf(val);
        }
    }
    __syncthreads();

    // ---- GEMM1: (16 x 448) x (448 x 256), bf16 WMMA, f32 accum ----
    v8f acc[NT1];
#pragma unroll
    for (int t = 0; t < NT1; ++t)
#pragma unroll
        for (int i = 0; i < 8; ++i) acc[t][i] = 0.0f;

#pragma unroll 1
    for (int ks = 0; ks < KS1; ++ks) {
        // A fragment (16-bit A 16x32 layout): row m = nlo,
        // e<8 -> k = 8*khalf + e ; e>=8 -> k = 16 + 8*khalf + (e-8)
        v8u alo = *(const v8u*)&rowbuf[nlo * FIN + 32 * ks + 8 * khalf];
        v8u ahi = *(const v8u*)&rowbuf[nlo * FIN + 32 * ks + 16 + 8 * khalf];
        BCast16 a;
#pragma unroll
        for (int i = 0; i < 8; ++i) { a.u[i] = alo[i]; a.u[8 + i] = ahi[i]; }

        const v16u* bp = (const v16u*)w1f + (size_t)ks * NT1 * 32;
#pragma unroll
        for (int t = 0; t < NT1; ++t) {
            BCast16 bfr;
            bfr.u = bp[t * 32 + lane];
            acc[t] = __builtin_amdgcn_wmma_f32_16x16x32_bf16(
                false, a.b, false, bfr.b, (short)0, acc[t], false, false);
        }
    }
    __syncthreads();

    // ---- bias + ReLU, write h (16x256 bf16) to LDS in row-major ----
    // C/D layout: lane holds n = nlo, VGPR r holds m = r + 8*khalf.
    unsigned short* hbuf = rowbuf;   // alias (rowbuf no longer needed)
#pragma unroll
    for (int t = 0; t < NT1; ++t) {
        float bb = b1[16 * t + nlo];
#pragma unroll
        for (int r = 0; r < 8; ++r) {
            float hv = fmaxf(acc[t][r] + bb, 0.0f);
            int m = r + 8 * khalf;
            hbuf[m * HID + 16 * t + nlo] = f2bf(hv);
        }
    }
    __syncthreads();

    // ---- GEMM2: (16 x 256) x (256 x 16) ----
    v8f acc2;
#pragma unroll
    for (int i = 0; i < 8; ++i) acc2[i] = 0.0f;

#pragma unroll
    for (int ks = 0; ks < KS2; ++ks) {
        v8u alo = *(const v8u*)&hbuf[nlo * HID + 32 * ks + 8 * khalf];
        v8u ahi = *(const v8u*)&hbuf[nlo * HID + 32 * ks + 16 + 8 * khalf];
        BCast16 a;
#pragma unroll
        for (int i = 0; i < 8; ++i) { a.u[i] = alo[i]; a.u[8 + i] = ahi[i]; }
        BCast16 bfr;
        bfr.u = *((const v16u*)w2f + ks * 32 + lane);
        acc2 = __builtin_amdgcn_wmma_f32_16x16x32_bf16(
            false, a.b, false, bfr.b, (short)0, acc2, false, false);
    }

    // ---- bias + store (f32 out, row-major (B*V, 16)) ----
    float bb2 = b2[nlo];
#pragma unroll
    for (int r = 0; r < 8; ++r) {
        int m = r + 8 * khalf;
        out[(size_t)(rowbase + m) * NOUT + nlo] = acc2[r] + bb2;
    }
}

// ---------------------------------------------------------------
extern "C" void kernel_launch(void* const* d_in, const int* in_sizes, int n_in,
                              void* d_out, int out_size, void* d_ws, size_t ws_size,
                              hipStream_t stream) {
    (void)in_sizes; (void)n_in; (void)out_size; (void)ws_size;
    const float* patch_tokens   = (const float*)d_in[0];
    const float* voxel_features = (const float*)d_in[1];
    const float* voxel_coords   = (const float*)d_in[2];
    const float* Kmat           = (const float*)d_in[3];
    const float* Rt             = (const float*)d_in[4];
    const float* W1             = (const float*)d_in[5];
    const float* b1             = (const float*)d_in[6];
    const float* W2             = (const float*)d_in[7];
    const float* b2             = (const float*)d_in[8];
    float* out = (float*)d_out;

    unsigned short* w1f = (unsigned short*)d_ws;                 // 229376 B
    unsigned short* w2f = w1f + (size_t)KS1 * NT1 * 32 * 16;     //   8192 B

    build_w1frag<<<(KS1 * NT1 * 32 * 16 + 255) / 256, 256, 0, stream>>>(W1, w1f);
    build_w2frag<<<(KS2 * 32 * 16 + 255) / 256, 256, 0, stream>>>(W2, w2f);

    const int rows = BB * VV;          // 131072
    fused_fusion_mlp<<<rows / 64, 128, 0, stream>>>(
        patch_tokens, voxel_features, voxel_coords, Kmat, Rt,
        w1f, b1, w2f, b2, out);
}
